// FlexibleAttention_51479478010130
// MI455X (gfx1250) — compile-verified, matrix operand order checked by reading, not run on hardware
//
#include <hip/hip_runtime.h>

// ---------------------------------------------------------------------------
// FlexibleAttention for MI455X (gfx1250, wave32, WMMA + TDM)
//
//   cvt: bf16 copies of X, Wo, W1, Wy (one conversion per element)
//   h_o = X@Wo^T, h_1 = X@W1^T                 (WMMA bf16, f32 out)
//   X_y = X@Wy^T + by                          (WMMA bf16, stored transposed bf16)
//   e_ij = tanh(h_o[i]+h_1[j]); s_ij = e.Wphi  (VALU, f32)
//   alpha = softmax_j(s)  (unnormalized exp; /l folded into epilogue)
//   part1 = alpha @ X_y                        (WMMA bf16, f32 acc)
//   g_i   = sum_j alpha_ij e_ij                (VALU, f32)
//   out   = (part1 + g@We^T)/l + be
// ---------------------------------------------------------------------------

typedef __bf16 bf16_t;
typedef __attribute__((ext_vector_type(16))) __bf16        v16bf;
typedef __attribute__((ext_vector_type(8)))  float         v8f;
typedef __attribute__((ext_vector_type(4)))  unsigned int  v4u;
typedef __attribute__((ext_vector_type(8)))  int           v8i;
typedef __attribute__((ext_vector_type(4)))  int           v4i;

#define HID   128
#define ADIM  64
#define BATCH 2
#define SEQ   512
#define MTOT  (BATCH * SEQ)   // 1024 total rows

// round-to-nearest-even f32 -> bf16
static __device__ __forceinline__ bf16_t f2bf(float f) {
  union { float f; unsigned int u; } x; x.f = f;
  unsigned int r = x.u + 0x7FFFu + ((x.u >> 16) & 1u);
  union { unsigned short s; bf16_t b; } y; y.s = (unsigned short)(r >> 16);
  return y.b;
}

// Two aligned 8x-bf16 chunks -> one 16x32 A/B-style fragment (ISA 7.12.2).
union frag_u {
  v16bf v;
  uint4 q[2];
};

static __device__ __forceinline__ v16bf load_frag_bf16(const bf16_t* __restrict__ row,
                                                       int k0, int k1) {
  frag_u u;
  u.q[0] = *(const uint4*)(row + k0);  // elems 0..7  : K = k0..k0+7
  u.q[1] = *(const uint4*)(row + k1);  // elems 8..15 : K = k1..k1+7
  return u.v;
}

// ---------------------------------------------------------------------------
// f32 -> bf16 bulk conversion (one pass, grid-stride)
// ---------------------------------------------------------------------------
__global__ __launch_bounds__(256)
void cvt_bf16_kernel(const float* __restrict__ src, bf16_t* __restrict__ dst, int n) {
  for (int i = blockIdx.x * 256 + threadIdx.x; i < n; i += gridDim.x * 256)
    dst[i] = f2bf(src[i]);
}

// ---------------------------------------------------------------------------
// Stage 1: C[M x Nout] = Xb[M x 128] @ Wb[Nout x 128]^T (+bias), bf16 inputs.
// One wave per 16x16 tile, 8 waves / block. Optionally writes f32 row-major
// and/or bf16 *transposed* [Nout x MTOT] (for stage-2 B-fragments).
// ---------------------------------------------------------------------------
__global__ __launch_bounds__(256)
void proj_gemm_kernel(const bf16_t* __restrict__ Xb, const bf16_t* __restrict__ Wb,
                      const float* __restrict__ bias, float* __restrict__ outF32,
                      bf16_t* __restrict__ outT, int Nout) {
  const int lane  = threadIdx.x & 31;
  const int wave  = threadIdx.x >> 5;
  const int tile  = blockIdx.x * 8 + wave;
  const int ntn   = Nout >> 4;
  const int mtile = tile / ntn;
  const int ntile = tile % ntn;
  const int nl    = lane & 15;
  const int sel   = lane >> 4;

  const bf16_t* xrow = Xb + (mtile * 16 + nl) * HID;  // A-frag: lane owns row nl
  const bf16_t* wrow = Wb + (ntile * 16 + nl) * HID;  // B-frag: lane owns col nl

  v8f c = {0.f, 0.f, 0.f, 0.f, 0.f, 0.f, 0.f, 0.f};
#pragma unroll
  for (int kb = 0; kb < HID; kb += 32) {
    const int k0 = kb + 8 * sel;
    const int k1 = kb + 16 + 8 * sel;
    v16bf a = load_frag_bf16(xrow, k0, k1);
    v16bf b = load_frag_bf16(wrow, k0, k1);
    c = __builtin_amdgcn_wmma_f32_16x16x32_bf16(false, a, false, b, (short)0, c,
                                                false, false);
  }

  const int n    = ntile * 16 + nl;
  const float bv = bias ? bias[n] : 0.0f;
  const int moff = sel * 8;
#pragma unroll
  for (int v = 0; v < 8; ++v) {
    const int mrow  = mtile * 16 + moff + v;  // C layout: lane col nl, rows 8*sel+v
    const float val = c[v] + bv;
    if (outF32) outF32[mrow * Nout + n] = val;
    if (outT)   outT[n * MTOT + mrow]   = f2bf(val);
  }
}

// ---------------------------------------------------------------------------
// Stage 2: one workgroup (8 waves) per 16-row i-tile (64 workgroups total)
// ---------------------------------------------------------------------------
__global__ __launch_bounds__(256)
void attn_kernel(const float* __restrict__ h_o, const float* __restrict__ h_1,
                 const bf16_t* __restrict__ XyT, const float* __restrict__ Wphi,
                 const float* __restrict__ We, const float* __restrict__ be,
                 float* __restrict__ out) {
  __shared__ float  s_p[16][SEQ];     // scores -> exp weights f32 (32 KB)
  __shared__ bf16_t s_pb[16][SEQ];    // exp weights bf16, WMMA A source (16 KB)
  __shared__ float  s_ho[16][ADIM];   // 4 KB (TDM target)
  __shared__ float  s_g[16][ADIM];
  __shared__ float  s_acc[16][HID];
  __shared__ float  s_wphi[ADIM];
  __shared__ float  s_rowmax[16];
  __shared__ float  s_rowsum[16];

  const int tid = threadIdx.x;
  const int b   = blockIdx.x >> 5;     // 32 i-tiles per batch
  const int it  = blockIdx.x & 31;
  const int i0  = it * 16;
  const int gro = b * SEQ;             // global row base of this batch

  // warm L2/WGP$ for the epilogue weights
  __builtin_prefetch(We + ((tid & 127) << 6), 0, 0);

  // phase 0: stage h_o tile (TDM DMA: contiguous 16*64 f32 tile) and Wphi
#if __has_builtin(__builtin_amdgcn_tensor_load_to_lds)
  if (tid < 32) {
    const unsigned long long ga =
        (unsigned long long)(const void*)(h_o + (size_t)(gro + i0) * ADIM);
    const unsigned lds_off = (unsigned)(unsigned long long)(void*)&s_ho[0][0];
    // D# group 0: count=1 | lds_addr | global_addr | type=2 (ISA 8.3)
    v4u g0;
    g0[0] = 1u;
    g0[1] = lds_off;
    g0[2] = (unsigned)ga;
    g0[3] = (unsigned)((ga >> 32) & 0x01FFFFFFu) | 0x80000000u;
    // D# group 1: data_size=4B, 1-D tile: tensor_dim0=tile_dim0=1024 (ISA 8.4)
    v8i g1;
    g1[0] = 0x00020000;                       // mask=0, data_size=2 (4 bytes)
    g1[1] = (int)((1024u & 0xFFFFu) << 16);   // tensor_dim0[15:0]
    g1[2] = (int)((1024u >> 16) | (1u << 16));// tensor_dim0[31:16], tensor_dim1=1
    g1[3] = (int)((1024u & 0xFFFFu) << 16);   // tensor_dim1[31:16]=0, tile_dim0=1024
    g1[4] = 1;                                // tile_dim1=1, tile_dim2=0
    g1[5] = 1024;                             // tensor_dim0_stride[31:0]
    g1[6] = 0;                                // stride0[47:32]=0, stride1[15:0]=0
    g1[7] = 0;
    v4i g2 = {0, 0, 0, 0};
    v4i g3 = {0, 0, 0, 0};
    v8i g4 = {0, 0, 0, 0, 0, 0, 0, 0};        // extra group (clang-23 6-arg form)
    __builtin_amdgcn_tensor_load_to_lds(g0, g1, g2, g3, g4, 0);
    __builtin_amdgcn_s_wait_tensorcnt(0);
  }
#else
  for (int idx = tid; idx < 16 * ADIM; idx += 256) {
    const int i = idx >> 6, a = idx & 63;
    s_ho[i][a] = h_o[(gro + i0 + i) * ADIM + a];
  }
#endif
  if (tid < ADIM) s_wphi[tid] = Wphi[tid];
  __syncthreads();

  // phase 1: scores s_ij = sum_a tanh(ho[i,a]+h1[j,a]) * Wphi[a]
  {
    const int i = tid & 15;
    for (int j = tid >> 4; j < SEQ; j += 16) {
      const float* h1r = h_1 + (gro + j) * ADIM;
      float acc = 0.f;
#pragma unroll 4
      for (int a = 0; a < ADIM; ++a)
        acc += tanhf(s_ho[i][a] + h1r[a]) * s_wphi[a];
      s_p[i][j] = acc;
    }
  }
  __syncthreads();

  // phase 2: row max, exp (f32 + bf16 copies, unnormalized), row sum
  if (tid < 16) {
    float mx = -3.4e38f;
    for (int j = 0; j < SEQ; ++j) mx = fmaxf(mx, s_p[tid][j]);
    s_rowmax[tid] = mx;
  }
  __syncthreads();
  for (int idx = tid; idx < 16 * SEQ; idx += 256) {
    const int i = idx >> 9, j = idx & (SEQ - 1);
    const float p = __expf(s_p[i][j] - s_rowmax[i]);
    s_p[i][j]  = p;
    s_pb[i][j] = f2bf(p);
  }
  __syncthreads();
  if (tid < 16) {
    float sm = 0.f;
    for (int j = 0; j < SEQ; ++j) sm += s_p[tid][j];
    s_rowsum[tid] = sm;
  }
  __syncthreads();

  // phase 3: part1 = P(16x512) @ X_y(512x128) via WMMA; wave w owns h-tile w.
  // A frags come from s_pb via 16B ds loads; B frags from XyT via 16B vmem loads.
  {
    const int lane = tid & 31;
    const int wave = tid >> 5;
    const int nl   = lane & 15;
    const int sel  = lane >> 4;
    const bf16_t* brow = XyT + (wave * 16 + nl) * MTOT + gro;  // contiguous over j
    const bf16_t* prow = &s_pb[nl][0];                         // A-frag row

    v8f c = {0.f, 0.f, 0.f, 0.f, 0.f, 0.f, 0.f, 0.f};
#pragma unroll 4
    for (int kb = 0; kb < SEQ; kb += 32) {
      const int k0 = kb + 8 * sel;
      const int k1 = kb + 16 + 8 * sel;
      v16bf a  = load_frag_bf16(prow, k0, k1);
      v16bf bb = load_frag_bf16(brow, k0, k1);
      c = __builtin_amdgcn_wmma_f32_16x16x32_bf16(false, a, false, bb, (short)0, c,
                                                  false, false);
    }
    const int moff = sel * 8;
#pragma unroll
    for (int v = 0; v < 8; ++v)
      s_acc[moff + v][wave * 16 + nl] = c[v];
  }
  __syncthreads();

  // phase 4: g[i,a] = sum_j p_ij * tanh(ho[i,a]+h1[j,a]); thread owns (i, 4 a's)
  {
    const int i  = tid >> 4;
    const int a0 = (tid & 15) * 4;
    const float ho0 = s_ho[i][a0 + 0], ho1 = s_ho[i][a0 + 1];
    const float ho2 = s_ho[i][a0 + 2], ho3 = s_ho[i][a0 + 3];
    float g0 = 0.f, g1 = 0.f, g2 = 0.f, g3 = 0.f;
    for (int j = 0; j < SEQ; ++j) {
      const float p = s_p[i][j];
      const float4 hv = *(const float4*)(h_1 + (gro + j) * ADIM + a0);
      g0 += p * tanhf(ho0 + hv.x);
      g1 += p * tanhf(ho1 + hv.y);
      g2 += p * tanhf(ho2 + hv.z);
      g3 += p * tanhf(ho3 + hv.w);
    }
    s_g[i][a0 + 0] = g0; s_g[i][a0 + 1] = g1;
    s_g[i][a0 + 2] = g2; s_g[i][a0 + 3] = g3;
  }
  __syncthreads();

  // phase 5: out = (part1 + g @ We^T) / l + be
  for (int idx = tid; idx < 16 * HID; idx += 256) {
    const int i = idx >> 7, h = idx & (HID - 1);
    float acc = s_acc[i][h];
    const float* wer = We + h * ADIM;
#pragma unroll 4
    for (int a = 0; a < ADIM; ++a) acc += wer[a] * s_g[i][a];
    out[(gro + i0 + i) * HID + h] = acc / s_rowsum[i] + be[h];
  }
}

// ---------------------------------------------------------------------------
extern "C" void kernel_launch(void* const* d_in, const int* in_sizes, int n_in,
                              void* d_out, int out_size, void* d_ws, size_t ws_size,
                              hipStream_t stream) {
  const float* X    = (const float*)d_in[0];
  const float* Wo   = (const float*)d_in[1];
  const float* W1   = (const float*)d_in[2];
  const float* Wphi = (const float*)d_in[3];
  const float* Wy   = (const float*)d_in[4];
  const float* by   = (const float*)d_in[5];
  const float* We   = (const float*)d_in[6];
  const float* be   = (const float*)d_in[7];
  float* out = (float*)d_out;

  // workspace layout
  char*   ws  = (char*)d_ws;
  float*  h_o = (float*)ws;                                   // 256 KB
  float*  h_1 = (float*)(ws + (size_t)MTOT * ADIM * 4);       // 256 KB
  bf16_t* XyT = (bf16_t*)(ws + (size_t)2 * MTOT * ADIM * 4);  // 256 KB
  char*   p   = ws + (size_t)2 * MTOT * ADIM * 4 + (size_t)HID * MTOT * 2;
  bf16_t* Xb  = (bf16_t*)p;                     p += (size_t)MTOT * HID * 2; // 256 KB
  bf16_t* Wob = (bf16_t*)p;                     p += (size_t)ADIM * HID * 2; // 16 KB
  bf16_t* W1b = (bf16_t*)p;                     p += (size_t)ADIM * HID * 2; // 16 KB
  bf16_t* Wyb = (bf16_t*)p;                                                  // 32 KB

  // one-shot bf16 conversions
  cvt_bf16_kernel<<<64, 256, 0, stream>>>(X,  Xb,  MTOT * HID);
  cvt_bf16_kernel<<<8,  256, 0, stream>>>(Wo, Wob, ADIM * HID);
  cvt_bf16_kernel<<<8,  256, 0, stream>>>(W1, W1b, ADIM * HID);
  cvt_bf16_kernel<<<8,  256, 0, stream>>>(Wy, Wyb, HID * HID);

  // h_o = X@Wo^T ; h_1 = X@W1^T  (64 x 4 = 256 tiles = 32 blocks)
  proj_gemm_kernel<<<32, 256, 0, stream>>>(Xb, Wob, nullptr, h_o, nullptr, ADIM);
  proj_gemm_kernel<<<32, 256, 0, stream>>>(Xb, W1b, nullptr, h_1, nullptr, ADIM);
  // X_y^T = (X@Wy^T + by)^T, bf16  (64 x 8 = 512 tiles = 64 blocks)
  proj_gemm_kernel<<<64, 256, 0, stream>>>(Xb, Wyb, by, nullptr, XyT, HID);
  // fused attention: 64 i-tiles
  attn_kernel<<<64, 256, 0, stream>>>(h_o, h_1, XyT, Wphi, We, be, out);
}
